// SpatialEdgeRNN_28381143892378
// MI455X (gfx1250) — matrix operand order repaired and checked by reading, not compiled, not run on hardware
//
#include <hip/hip_runtime.h>
#include <hip/hip_bf16.h>
#include <math.h>

// Shapes from the reference
#define SS 16
#define MM 64
#define HH 256
#define EE 64
#define KT (EE + HH)          // 320 concatenated K ( [e | ht] )
#define ROWS_PER_BLK 128      // pair rows per block (8 waves x 16 rows)
#define ASTRIDE 328           // padded LDS row stride in halves (164 dwords -> conflict-free)
#define NROWS_B 64            // 4 gates x 16 h-columns of weight panel
#define NPAIRS (SS * MM * MM) // 65536
#define SMEM_BYTES ((ROWS_PER_BLK + NROWS_B) * ASTRIDE * 2)

typedef __attribute__((ext_vector_type(16))) _Float16 v16h;
typedef __attribute__((ext_vector_type(8)))  _Float16 v8h;
typedef __attribute__((ext_vector_type(4)))  _Float16 v4h;
typedef __attribute__((ext_vector_type(8)))  float    v8f;
typedef __attribute__((ext_vector_type(4)))  float    v4f;

__device__ __forceinline__ v16h cat16(v8h lo, v8h hi) {
    v16h r;
#pragma unroll
    for (int t = 0; t < 8; ++t) { r[t] = lo[t]; r[t + 8] = hi[t]; }
    return r;
}

__device__ __forceinline__ float sigm(float x) {
    return 1.0f / (1.0f + __expf(-x));
}

__global__ __launch_bounds__(256)
void spatial_edge_lstm_wmma(const float* __restrict__ ht,
                            const float* __restrict__ ct,
                            const float* __restrict__ traj,
                            const int*   __restrict__ tsmask,
                            const float* __restrict__ We,
                            const float* __restrict__ be,
                            const float* __restrict__ Wih,
                            const float* __restrict__ Whh,
                            const float* __restrict__ bih,
                            const float* __restrict__ bhh,
                            float* __restrict__ out_ht,
                            float* __restrict__ out_ct) {
    extern __shared__ char smem[];
    _Float16* Alds = (_Float16*)smem;                                   // [128][328]
    _Float16* Blds = (_Float16*)(smem + ROWS_PER_BLK * ASTRIDE * 2);    // [64][328]
    __shared__ int exist_sh[MM];
    __shared__ int active_sh;

    const int tid     = threadIdx.x;
    const int rowBase = blockIdx.x * ROWS_PER_BLK;   // flat pair row base
    const int sIdx    = rowBase >> 12;               // 4096 pair rows per scene
    const int hoBase  = blockIdx.y * 16;             // hidden-column tile base

    // ---- existence mask for this scene ----
    if (tid < MM) exist_sh[tid] = (tsmask[sIdx * MM + tid] == 1) ? 1 : 0;

    // ---- weight panel: B[n][k], n = gate*16 + hcol, k = [e(64) | h(256)] ----
    for (int idx = tid; idx < NROWS_B * KT; idx += 256) {
        int n  = idx / KT;
        int k  = idx - n * KT;
        int gi = n >> 4;
        int g  = gi * HH + hoBase + (n & 15);        // row of [4H] weight matrix
        float w = (k < EE) ? Wih[g * EE + k] : Whh[g * HH + (k - EE)];
        Blds[n * ASTRIDE + k] = (_Float16)w;
    }

    // ---- ht tile -> A[:, 64:320] as f16 (b128 global loads, b64 LDS stores) ----
    for (int idx = tid; idx < ROWS_PER_BLK * (HH / 4); idx += 256) {
        int row = idx >> 6;
        int c4  = idx & 63;
        v4f v = *(const v4f*)(ht + (size_t)(rowBase + row) * HH + c4 * 4);
        v4h h4;
        h4[0] = (_Float16)v[0]; h4[1] = (_Float16)v[1];
        h4[2] = (_Float16)v[2]; h4[3] = (_Float16)v[3];
        *(v4h*)(Alds + row * ASTRIDE + EE + c4 * 4) = h4;
    }

    // ---- embed: e = relu(diff @ W_embed + b_embed) -> A[:, 0:64] ----
    {
        int row = tid >> 1;                          // 0..127
        int R   = rowBase + row;
        int i   = (R >> 6) & 63;
        int j   = R & 63;
        float dx = traj[(sIdx * MM + j) * 2 + 0] - traj[(sIdx * MM + i) * 2 + 0];
        float dy = traj[(sIdx * MM + j) * 2 + 1] - traj[(sIdx * MM + i) * 2 + 1];
        int k0 = (tid & 1) * 32;
#pragma unroll 8
        for (int k = k0; k < k0 + 32; ++k) {
            float ev = fmaf(dx, We[k], fmaf(dy, We[EE + k], be[k]));
            Alds[row * ASTRIDE + k] = (_Float16)fmaxf(ev, 0.0f);
        }
    }
    __syncthreads();

    if (tid == 0) {
        int c = 0;
#pragma unroll
        for (int k = 0; k < MM; ++k) c += exist_sh[k];
        active_sh = (c > 1) ? 1 : 0;
    }
    __syncthreads();

    // ---- WMMA: each wave owns 16 rows, accumulates 4 gate tiles ----
    const int lane = tid & 31;
    const int wv   = tid >> 5;
    const int l15  = lane & 15;
    const int lhi  = lane >> 4;                      // 0 or 1
    const int r0   = wv * 16;

    v8f acc0 = {}, acc1 = {}, acc2 = {}, acc3 = {};
    const _Float16* Arow = Alds + (r0 + l15) * ASTRIDE;

#pragma unroll
    for (int kb = 0; kb < KT / 32; ++kb) {
        // A fragment: halves 0..15 -> K = kb*32 + 16*lhi + t
        int ka = kb * 32 + 16 * lhi;
        v16h a = cat16(*(const v8h*)(Arow + ka), *(const v8h*)(Arow + ka + 8));
        // B fragment: lane = N column, 16 contiguous K halves per lane-half
        int kbb = kb * 32 + 16 * lhi;
        const _Float16* Bp = Blds + l15 * ASTRIDE + kbb;
        v16h b0 = cat16(*(const v8h*)(Bp + 0 * 16 * ASTRIDE),
                        *(const v8h*)(Bp + 0 * 16 * ASTRIDE + 8));
        v16h b1 = cat16(*(const v8h*)(Bp + 1 * 16 * ASTRIDE),
                        *(const v8h*)(Bp + 1 * 16 * ASTRIDE + 8));
        v16h b2 = cat16(*(const v8h*)(Bp + 2 * 16 * ASTRIDE),
                        *(const v8h*)(Bp + 2 * 16 * ASTRIDE + 8));
        v16h b3 = cat16(*(const v8h*)(Bp + 3 * 16 * ASTRIDE),
                        *(const v8h*)(Bp + 3 * 16 * ASTRIDE + 8));
        acc0 = __builtin_amdgcn_wmma_f32_16x16x32_f16(false, a, false, b0, (short)0, acc0, false, false);
        acc1 = __builtin_amdgcn_wmma_f32_16x16x32_f16(false, a, false, b1, (short)0, acc1, false, false);
        acc2 = __builtin_amdgcn_wmma_f32_16x16x32_f16(false, a, false, b2, (short)0, acc2, false, false);
        acc3 = __builtin_amdgcn_wmma_f32_16x16x32_f16(false, a, false, b3, (short)0, acc3, false, false);
    }

    // ---- epilogue: LSTM cell + masked blend ----
    const int h = hoBase + l15;                      // N = lane%16
    const float bs0 = bih[0 * HH + h] + bhh[0 * HH + h];
    const float bs1 = bih[1 * HH + h] + bhh[1 * HH + h];
    const float bs2 = bih[2 * HH + h] + bhh[2 * HH + h];
    const float bs3 = bih[3 * HH + h] + bhh[3 * HH + h];
    const int active = active_sh;

#pragma unroll
    for (int v = 0; v < 8; ++v) {
        int R = rowBase + r0 + v + 8 * lhi;          // C layout: M = v + 8*(lane>=16)
        int i = (R >> 6) & 63;
        int j = R & 63;
        float pm = (exist_sh[i] & exist_sh[j] & active) ? 1.0f : 0.0f;
        size_t g = (size_t)R * HH + h;
        float c_old = ct[g];
        float h_old = ht[g];
        float ig = sigm(acc0[v] + bs0);
        float fg = sigm(acc1[v] + bs1);
        float gg = tanhf(acc2[v] + bs2);
        float og = sigm(acc3[v] + bs3);
        float cn = fg * c_old + ig * gg;
        float hn = og * tanhf(cn);
        out_ht[g] = pm * hn + (1.0f - pm) * h_old;
        out_ct[g] = pm * cn + (1.0f - pm) * c_old;
    }
}

extern "C" void kernel_launch(void* const* d_in, const int* in_sizes, int n_in,
                              void* d_out, int out_size, void* d_ws, size_t ws_size,
                              hipStream_t stream) {
    const float* ht     = (const float*)d_in[0];
    const float* ct     = (const float*)d_in[1];
    const float* traj   = (const float*)d_in[2];
    const int*   tsmask = (const int*)d_in[3];
    const float* We     = (const float*)d_in[4];
    const float* be     = (const float*)d_in[5];
    const float* Wih    = (const float*)d_in[6];
    const float* Whh    = (const float*)d_in[7];
    const float* bih    = (const float*)d_in[8];
    const float* bhh    = (const float*)d_in[9];

    float* out_ht = (float*)d_out;
    float* out_ct = out_ht + (size_t)NPAIRS * HH;

    dim3 grid(NPAIRS / ROWS_PER_BLK, HH / 16, 1);    // 512 x 16
    dim3 block(256, 1, 1);
    spatial_edge_lstm_wmma<<<grid, block, SMEM_BYTES, stream>>>(
        ht, ct, traj, tsmask, We, be, Wih, Whh, bih, bhh, out_ht, out_ct);
}
